// Net_90013924590456
// MI455X (gfx1250) — matrix operand order repaired and checked by reading, not compiled, not run on hardware
//
#include <hip/hip_runtime.h>
#include <hip/hip_bf16.h>

// ---------------------------------------------------------------------------
// GIN + virtual-node GNN forward for MI455X (gfx1250, wave32, WMMA).
// All dense GEMMs use v_wmma_f32_16x16x32_f16 (f32 accumulate).
// Weights pre-transposed to f16 [NOUT][K] so each B fragment is one v16h load.
// ---------------------------------------------------------------------------

#define NN 262144
#define EE 1048576
#define GG 8192
#define HH 128

typedef __attribute__((ext_vector_type(16))) _Float16 v16h;
typedef __attribute__((ext_vector_type(8)))  _Float16 v8h;
typedef __attribute__((ext_vector_type(8)))  float    v8f;

enum { AMODE_PLAIN = 0, AMODE_GIN = 1, AMODE_CONCAT = 2 };
enum { EMODE_BIAS = 0, EMODE_BIAS_RELU = 1, EMODE_BN_RELU = 2, EMODE_BN_RELU_POOL = 3 };

// ---------------------------------------------------------------------------
// Generic tall-skinny GEMM: D[rows,NOUT] = epilogue(A[rows,K] @ W[K,NOUT]).
// Block = 256 threads = 8 waves; wave owns a 16-row strip; block = 128 rows.
// A fragments hoisted over n-tiles; B from f16 transposed weights (v16h load).
// Fragment layouts used here (contiguous k, same order for A and B):
//   A (16x32 f16): lane(m,half) holds k = kt*32 + half*16 + p*8 + i  (i=0..7)
//   B (32x16 f16): lane(n=m,half) holds k = kt*32 + half*16 + 0..15 (16 consec)
//   C/D:           lane(n=m,half) holds rows half*8 + j               (j=0..7)
// ---------------------------------------------------------------------------
template <int K, int NOUT, int AMODE, int EMODE, bool AF16, bool DF16>
__global__ __launch_bounds__(256) void wmma_gemm(
    const void*  __restrict__ A0,    // PLAIN: A (f32 or f16) | GIN: h | CONCAT: vn
    const float* __restrict__ A1,    // GIN: agg | CONCAT: h [rows,K/2]
    const float* __restrict__ epsp,  // GIN: &conv_eps[layer]
    const int*   __restrict__ batch, // CONCAT / POOL: [rows]
    const _Float16* __restrict__ Wt, // [NOUT, K] f16 (pre-transposed)
    const float* __restrict__ bias,  // [NOUT]
    const float* __restrict__ bng, const float* __restrict__ bnb,
    const float* __restrict__ bnm, const float* __restrict__ bnv,
    void* __restrict__ D,            // [rows,NOUT] f32/f16, or pooled (atomic f32)
    int rows)
{
    const int wave = threadIdx.x >> 5;
    const int lane = threadIdx.x & 31;
    const int half = lane >> 4;
    const int m    = lane & 15;
    const int rowBase = blockIdx.x * 128 + wave * 16;
    const int aRow = rowBase + m;
    (void)rows;

    constexpr int KT = K / 32;
    v16h a[KT];

    float epsf = 0.f;
    if constexpr (AMODE == AMODE_GIN) epsf = 1.0f + epsp[0];
    int abatch = 0;
    if constexpr (AMODE == AMODE_CONCAT) abatch = batch[aRow];

    // ---- load & convert A fragments: two 8-wide consecutive runs per ktile ----
#pragma unroll
    for (int kt = 0; kt < KT; ++kt) {
#pragma unroll
        for (int p = 0; p < 2; ++p) {
            const int kb = kt * 32 + half * 16 + p * 8;  // 8 consecutive k
            if constexpr (AMODE == AMODE_PLAIN && AF16) {
                const v8h t = *(const v8h*)((const _Float16*)A0 + (size_t)aRow * K + kb);
#pragma unroll
                for (int i = 0; i < 8; ++i) a[kt][p * 8 + i] = t[i];
            } else if constexpr (AMODE == AMODE_PLAIN) {
                const float* ap = (const float*)A0 + (size_t)aRow * K + kb;
                const float4 f0 = *(const float4*)(ap);
                const float4 f1 = *(const float4*)(ap + 4);
                a[kt][p*8+0] = (_Float16)f0.x; a[kt][p*8+1] = (_Float16)f0.y;
                a[kt][p*8+2] = (_Float16)f0.z; a[kt][p*8+3] = (_Float16)f0.w;
                a[kt][p*8+4] = (_Float16)f1.x; a[kt][p*8+5] = (_Float16)f1.y;
                a[kt][p*8+6] = (_Float16)f1.z; a[kt][p*8+7] = (_Float16)f1.w;
            } else if constexpr (AMODE == AMODE_GIN) {
                const float* hp = (const float*)A0 + (size_t)aRow * K + kb;
                const float* gp = A1 + (size_t)aRow * K + kb;
                const float4 h0 = *(const float4*)(hp);
                const float4 h1 = *(const float4*)(hp + 4);
                const float4 g0 = *(const float4*)(gp);
                const float4 g1 = *(const float4*)(gp + 4);
                a[kt][p*8+0] = (_Float16)(epsf * h0.x + g0.x);
                a[kt][p*8+1] = (_Float16)(epsf * h0.y + g0.y);
                a[kt][p*8+2] = (_Float16)(epsf * h0.z + g0.z);
                a[kt][p*8+3] = (_Float16)(epsf * h0.w + g0.w);
                a[kt][p*8+4] = (_Float16)(epsf * h1.x + g1.x);
                a[kt][p*8+5] = (_Float16)(epsf * h1.y + g1.y);
                a[kt][p*8+6] = (_Float16)(epsf * h1.z + g1.z);
                a[kt][p*8+7] = (_Float16)(epsf * h1.w + g1.w);
            } else { // CONCAT: run of 8 never straddles the K/2 boundary
                constexpr int Hc = K / 2;
                const float* sp = (kb < Hc)
                    ? ((const float*)A0 + (size_t)abatch * Hc + kb)
                    : (A1 + (size_t)aRow * Hc + (kb - Hc));
                const float4 f0 = *(const float4*)(sp);
                const float4 f1 = *(const float4*)(sp + 4);
                a[kt][p*8+0] = (_Float16)f0.x; a[kt][p*8+1] = (_Float16)f0.y;
                a[kt][p*8+2] = (_Float16)f0.z; a[kt][p*8+3] = (_Float16)f0.w;
                a[kt][p*8+4] = (_Float16)f1.x; a[kt][p*8+5] = (_Float16)f1.y;
                a[kt][p*8+6] = (_Float16)f1.z; a[kt][p*8+7] = (_Float16)f1.w;
            }
        }
    }

    int prow[8];
    if constexpr (EMODE == EMODE_BN_RELU_POOL) {
#pragma unroll
        for (int j = 0; j < 8; ++j) prow[j] = batch[rowBase + half * 8 + j];
    }

#pragma unroll 1
    for (int nt = 0; nt < NOUT / 16; ++nt) {
        const int n = nt * 16 + m;
        const _Float16* wrow = Wt + (size_t)n * K;
        v8f c = {};
#pragma unroll
        for (int kt = 0; kt < KT; ++kt) {
            const v16h b = *(const v16h*)(wrow + kt * 32 + half * 16);
            c = __builtin_amdgcn_wmma_f32_16x16x32_f16(
                    false, a[kt], false, b, (short)0, c, false, false);
        }
        // epilogue: fold bias (+BatchNorm) into per-channel scale/shift
        float scale = 1.f, shift = bias[n];
        if constexpr (EMODE >= EMODE_BN_RELU) {
            const float s = bng[n] * __frsqrt_rn(bnv[n] + 1e-5f);
            scale = s;
            shift = (bias[n] - bnm[n]) * s + bnb[n];
        }
#pragma unroll
        for (int j = 0; j < 8; ++j) {
            float val = c[j] * scale + shift;
            if constexpr (EMODE != EMODE_BIAS) val = fmaxf(val, 0.f);
            if constexpr (EMODE == EMODE_BN_RELU_POOL) {
                __hip_atomic_fetch_add((float*)D + (size_t)prow[j] * NOUT + n, val,
                                       __ATOMIC_RELAXED, __HIP_MEMORY_SCOPE_AGENT);
            } else if constexpr (DF16) {
                ((_Float16*)D)[(size_t)(rowBase + half * 8 + j) * NOUT + n] = (_Float16)val;
            } else {
                ((float*)D)[(size_t)(rowBase + half * 8 + j) * NOUT + n] = val;
            }
        }
    }
}

// ---------------------------------------------------------------------------
// Weight prep: W[K,NOUT] f32 -> Wt[NOUT,K] f16 (coalesced read, scattered write)
// ---------------------------------------------------------------------------
__global__ __launch_bounds__(256) void wt_prep_kernel(
    const float* __restrict__ W, _Float16* __restrict__ Wt, int K, int NOUT)
{
    const int i = blockIdx.x * 256 + threadIdx.x;
    if (i >= K * NOUT) return;
    const int k = i / NOUT, n = i % NOUT;
    Wt[(size_t)n * K + k] = (_Float16)W[i];
}

// ---------------------------------------------------------------------------
// AtomEncoder: h[n,c] = sum_f atom_emb[f, x[n,f], c]. One 128-thread block/node.
// ---------------------------------------------------------------------------
__global__ __launch_bounds__(128) void atom_encoder_kernel(
    const int* __restrict__ x, const float* __restrict__ emb,
    float* __restrict__ h)
{
    const int n = blockIdx.x;
    const int c = threadIdx.x;
    __shared__ int xi[9];
    if (c < 9) xi[c] = x[n * 9 + c];
    __syncthreads();
    float s = 0.f;
#pragma unroll
    for (int f = 0; f < 9; ++f) s += emb[((size_t)f * 64 + xi[f]) * HH + c];
    h[(size_t)n * HH + c] = s;
}

__global__ __launch_bounds__(256) void vn_init_kernel(
    float* __restrict__ vn, const float* __restrict__ vn_emb, int total4)
{
    const int i = blockIdx.x * 256 + threadIdx.x;
    if (i >= total4) return;
    ((float4*)vn)[i] = *(const float4*)(vn_emb + (i & 31) * 4);
}

__global__ __launch_bounds__(256) void add_vn_kernel(
    float* __restrict__ h, const float* __restrict__ vn,
    const int* __restrict__ batch, int total4)
{
    const int i = blockIdx.x * 256 + threadIdx.x;
    if (i >= total4) return;
    const int n = i >> 5, c4 = (i & 31) * 4;
    const int g = batch[n];
    float4 hv = *(float4*)(h + (size_t)n * HH + c4);
    const float4 vv = *(const float4*)(vn + (size_t)g * HH + c4);
    hv.x += vv.x; hv.y += vv.y; hv.z += vv.z; hv.w += vv.w;
    *(float4*)(h + (size_t)n * HH + c4) = hv;
}

__global__ __launch_bounds__(256) void zero_kernel(float* __restrict__ p, size_t n4)
{
    const size_t i = (size_t)blockIdx.x * 256 + threadIdx.x;
    if (i < n4) ((float4*)p)[i] = make_float4(0.f, 0.f, 0.f, 0.f);
}

// ---------------------------------------------------------------------------
// Edge message + scatter: wave per edge, lane = 4 channels (float4).
// msg = relu(h[src] + sum_f bond_emb[f, eattr[e,f]]); agg[dst] += msg (atomic).
// ---------------------------------------------------------------------------
__global__ __launch_bounds__(256) void edge_msg_kernel(
    const int* __restrict__ srcv, const int* __restrict__ dstv,
    const int* __restrict__ eattr, const float* __restrict__ bemb, // [3][8][128]
    const float* __restrict__ h, float* __restrict__ agg, int E)
{
    const int e = blockIdx.x * 8 + (threadIdx.x >> 5);
    const int lane = threadIdx.x & 31;
    if (e >= E) return;
    const int s = srcv[e], d = dstv[e];
    const int a0 = eattr[e * 3 + 0], a1 = eattr[e * 3 + 1], a2 = eattr[e * 3 + 2];
    const int c = lane * 4;
    const float4 hv = *(const float4*)(h + (size_t)s * HH + c);
    const float4 e0 = *(const float4*)(bemb + (size_t)(0 * 8 + a0) * HH + c);
    const float4 e1 = *(const float4*)(bemb + (size_t)(1 * 8 + a1) * HH + c);
    const float4 e2 = *(const float4*)(bemb + (size_t)(2 * 8 + a2) * HH + c);
    float m0 = fmaxf(hv.x + e0.x + e1.x + e2.x, 0.f);
    float m1 = fmaxf(hv.y + e0.y + e1.y + e2.y, 0.f);
    float m2 = fmaxf(hv.z + e0.z + e1.z + e2.z, 0.f);
    float m3 = fmaxf(hv.w + e0.w + e1.w + e2.w, 0.f);
    float* out = agg + (size_t)d * HH + c;
    __hip_atomic_fetch_add(out + 0, m0, __ATOMIC_RELAXED, __HIP_MEMORY_SCOPE_AGENT);
    __hip_atomic_fetch_add(out + 1, m1, __ATOMIC_RELAXED, __HIP_MEMORY_SCOPE_AGENT);
    __hip_atomic_fetch_add(out + 2, m2, __ATOMIC_RELAXED, __HIP_MEMORY_SCOPE_AGENT);
    __hip_atomic_fetch_add(out + 3, m3, __ATOMIC_RELAXED, __HIP_MEMORY_SCOPE_AGENT);
}

// ---------------------------------------------------------------------------
// Final head: out[g] = clip(relu_p[g,:] . pW2 + pb2, 0, 50). Wave per graph.
// ---------------------------------------------------------------------------
__global__ __launch_bounds__(256) void final_dot_kernel(
    const float* __restrict__ p, const float* __restrict__ w2,
    const float* __restrict__ b2, float* __restrict__ out, int G)
{
    const int g = blockIdx.x * 8 + (threadIdx.x >> 5);
    const int lane = threadIdx.x & 31;
    if (g >= G) return;
    const int c = lane * 4;
    const float4 pv = *(const float4*)(p + (size_t)g * HH + c);
    const float4 wv = *(const float4*)(w2 + c);
    float s = pv.x * wv.x + pv.y * wv.y + pv.z * wv.z + pv.w * wv.w;
#pragma unroll
    for (int off = 16; off > 0; off >>= 1) s += __shfl_down(s, off, 32);
    if (lane == 0) {
        float v = s + b2[0];
        out[g] = fminf(fmaxf(v, 0.f), 50.f);
    }
}

// ---------------------------------------------------------------------------
extern "C" void kernel_launch(void* const* d_in, const int* in_sizes, int n_in,
                              void* d_out, int out_size, void* d_ws, size_t ws_size,
                              hipStream_t stream)
{
    (void)in_sizes; (void)n_in; (void)out_size; (void)ws_size;
    const int*   x          = (const int*)d_in[0];
    const int*   edge_index = (const int*)d_in[1];
    const int*   eattr      = (const int*)d_in[2];
    const int*   batch      = (const int*)d_in[3];
    const float* atom_emb   = (const float*)d_in[4];
    const float* vn_emb     = (const float*)d_in[5];
    const float* bond_emb   = (const float*)d_in[6];
    const float* conv_eps   = (const float*)d_in[7];
    const float* conv_W1    = (const float*)d_in[8];
    const float* conv_b1    = (const float*)d_in[9];
    const float* conv_bn_g  = (const float*)d_in[10];
    const float* conv_bn_b  = (const float*)d_in[11];
    const float* conv_bn_m  = (const float*)d_in[12];
    const float* conv_bn_v  = (const float*)d_in[13];
    const float* conv_W2    = (const float*)d_in[14];
    const float* conv_b2    = (const float*)d_in[15];
    const float* vn1_W      = (const float*)d_in[16];
    const float* vn1_b      = (const float*)d_in[17];
    const float* vn1_g      = (const float*)d_in[18];
    const float* vn1_be     = (const float*)d_in[19];
    const float* vn1_m      = (const float*)d_in[20];
    const float* vn1_v      = (const float*)d_in[21];
    const float* vn2_W      = (const float*)d_in[22];
    const float* vn2_b      = (const float*)d_in[23];
    const float* vn2_g      = (const float*)d_in[24];
    const float* vn2_be     = (const float*)d_in[25];
    const float* vn2_m      = (const float*)d_in[26];
    const float* vn2_v      = (const float*)d_in[27];
    const float* pW1        = (const float*)d_in[28];
    const float* pb1        = (const float*)d_in[29];
    const float* pW2        = (const float*)d_in[30];
    const float* pb2        = (const float*)d_in[31];

    // ---- workspace carve: f32 region then f16 region ----
    float* ws = (float*)d_ws;
    size_t off = 0;
    float* hA     = ws + off; off += (size_t)NN * HH;
    float* hB     = ws + off; off += (size_t)NN * HH;
    float* agg    = ws + off; off += (size_t)NN * HH;
    float* pooled = ws + off; off += (size_t)GG * 2 * HH;
    float* vn     = ws + off; off += (size_t)GG * HH;
    float* pbuf   = ws + off; off += (size_t)GG * HH;
    _Float16* hws = (_Float16*)(ws + off);
    size_t hoff = 0;
    _Float16* z2h   = hws + hoff; hoff += (size_t)NN * 2 * HH;
    _Float16* w1t   = hws + hoff; hoff += (size_t)4 * 128 * 256;  // per-layer [256,128]
    _Float16* w2t   = hws + hoff; hoff += (size_t)4 * 256 * 128;  // per-layer [128,256]
    _Float16* vn1t  = hws + hoff; hoff += (size_t)4 * 256 * 256;
    _Float16* vn2t  = hws + hoff; hoff += (size_t)4 * 256 * 128;
    _Float16* pw1t  = hws + hoff; hoff += (size_t)128 * 128;

    const int* srcv = edge_index;
    const int* dstv = edge_index + EE;

    // ---- weight prep: transpose + f32->f16 (tiny; L2-resident thereafter) ----
    for (int i = 0; i < 4; ++i) {
        wt_prep_kernel<<<(128 * 256 + 255) / 256, 256, 0, stream>>>(
            conv_W1 + (size_t)i * 128 * 256, w1t + (size_t)i * 128 * 256, 128, 256);
        wt_prep_kernel<<<(256 * 128 + 255) / 256, 256, 0, stream>>>(
            conv_W2 + (size_t)i * 256 * 128, w2t + (size_t)i * 256 * 128, 256, 128);
        wt_prep_kernel<<<(256 * 256 + 255) / 256, 256, 0, stream>>>(
            vn1_W + (size_t)i * 256 * 256, vn1t + (size_t)i * 256 * 256, 256, 256);
        wt_prep_kernel<<<(256 * 128 + 255) / 256, 256, 0, stream>>>(
            vn2_W + (size_t)i * 256 * 128, vn2t + (size_t)i * 256 * 128, 256, 128);
    }
    wt_prep_kernel<<<(128 * 128 + 255) / 256, 256, 0, stream>>>(pW1, pw1t, 128, 128);

    // h0 = AtomEncoder(x); vn0 = broadcast(vn_emb)
    atom_encoder_kernel<<<NN, 128, 0, stream>>>(x, atom_emb, hA);
    {
        const int t4 = GG * HH / 4;
        vn_init_kernel<<<(t4 + 255) / 256, 256, 0, stream>>>(vn, vn_emb, t4);
    }

    float* hcur = hA;
    float* hnext = hB;
    for (int i = 0; i < 4; ++i) {
        // h += vn[batch]
        {
            const int t4 = NN * HH / 4;
            add_vn_kernel<<<(t4 + 255) / 256, 256, 0, stream>>>(hcur, vn, batch, t4);
        }
        // agg = segment_sum(relu(h[src] + bond_emb), dst)
        {
            const size_t n4 = (size_t)NN * HH / 4;
            zero_kernel<<<(unsigned)((n4 + 255) / 256), 256, 0, stream>>>(agg, n4);
        }
        edge_msg_kernel<<<EE / 8, 256, 0, stream>>>(
            srcv, dstv, eattr, bond_emb + (size_t)i * 3 * 8 * HH, hcur, agg, EE);

        // z2 = relu(BN(((1+eps)h + agg) @ W1 + b1))   [N,256] f16
        wmma_gemm<128, 256, AMODE_GIN, EMODE_BN_RELU, false, true>
            <<<NN / 128, 256, 0, stream>>>(
            hcur, agg, conv_eps + i, nullptr,
            w1t + (size_t)i * 128 * 256, conv_b1 + (size_t)i * 256,
            conv_bn_g + (size_t)i * 256, conv_bn_b + (size_t)i * 256,
            conv_bn_m + (size_t)i * 256, conv_bn_v + (size_t)i * 256, z2h, NN);

        // hnext = z2 @ W2 + b2   [N,128] f32
        wmma_gemm<256, 128, AMODE_PLAIN, EMODE_BIAS, true, false>
            <<<NN / 128, 256, 0, stream>>>(
            z2h, nullptr, nullptr, nullptr,
            w2t + (size_t)i * 256 * 128, conv_b2 + (size_t)i * 128,
            nullptr, nullptr, nullptr, nullptr, hnext, NN);

        // pooled = segment_sum(relu(BN(concat(vn[batch],hnext) @ vn1_W + b)), batch)
        {
            const size_t n4 = (size_t)GG * 2 * HH / 4;
            zero_kernel<<<(unsigned)((n4 + 255) / 256), 256, 0, stream>>>(pooled, n4);
        }
        wmma_gemm<256, 256, AMODE_CONCAT, EMODE_BN_RELU_POOL, false, false>
            <<<NN / 128, 256, 0, stream>>>(
            vn, hnext, nullptr, batch,
            vn1t + (size_t)i * 256 * 256, vn1_b + (size_t)i * 256,
            vn1_g + (size_t)i * 256, vn1_be + (size_t)i * 256,
            vn1_m + (size_t)i * 256, vn1_v + (size_t)i * 256, pooled, NN);

        // vn = relu(BN(pooled @ vn2_W + vn2_b))   [G,128]
        wmma_gemm<256, 128, AMODE_PLAIN, EMODE_BN_RELU, false, false>
            <<<GG / 128, 256, 0, stream>>>(
            pooled, nullptr, nullptr, nullptr,
            vn2t + (size_t)i * 256 * 128, vn2_b + (size_t)i * 128,
            vn2_g + (size_t)i * 128, vn2_be + (size_t)i * 128,
            vn2_m + (size_t)i * 128, vn2_v + (size_t)i * 128, vn, GG);

        float* t = hcur; hcur = hnext; hnext = t;
    }

    // head: pbuf = relu(vn @ pW1 + pb1); out = clip(pbuf @ pW2 + pb2, 0, 50)
    wmma_gemm<128, 128, AMODE_PLAIN, EMODE_BIAS_RELU, false, false>
        <<<GG / 128, 256, 0, stream>>>(
        vn, nullptr, nullptr, nullptr, pw1t, pb1,
        nullptr, nullptr, nullptr, nullptr, pbuf, GG);
    final_dot_kernel<<<GG / 8, 256, 0, stream>>>(pbuf, pW2, pb2, (float*)d_out, GG);
}
